// HeadAttention_68959994904762
// MI455X (gfx1250) — matrix operand order repaired and checked by reading, not compiled, not run on hardware
//
#include <hip/hip_runtime.h>
#include <hip/hip_bf16.h>

// ---------- types ----------
typedef __bf16 bf16;
typedef __attribute__((ext_vector_type(16))) __bf16 v16bf;
typedef __attribute__((ext_vector_type(8)))  __bf16 v8bf;
typedef __attribute__((ext_vector_type(8)))  float  v8f;
typedef __attribute__((address_space(3)))    bf16   lds_bf16_t;

#define BATCH 8
#define SEQ   2048
#define EMBD  1024
#define HEADD 128
#define ROWS  (BATCH * SEQ)      // 16384
#define NEGBIG (-1.0e30f)

// Load a 16-element bf16 fragment as two contiguous 16B chunks (-> ds/global b128).
static __device__ __forceinline__ v16bf frag16(const bf16* p0, const bf16* p1) {
    v8bf a = *(const v8bf*)p0;
    v8bf b = *(const v8bf*)p1;
    return __builtin_shufflevector(a, b, 0,1,2,3,4,5,6,7,8,9,10,11,12,13,14,15);
}

static __device__ __forceinline__ v8f wmma_bf16(v16bf a, v16bf b, v8f c) {
    // (neg_a, A, neg_b, B, c_mod, C, reuse_a, reuse_b)
    return __builtin_amdgcn_wmma_f32_16x16x32_bf16(false, a, false, b, (short)0, c, false, false);
}

// CDNA5 async DMA: global -> LDS, 16 bytes per lane, tracked on ASYNCcnt.
static __device__ __forceinline__ void async_ld_b128(bf16* lptr, const bf16* gptr) {
    unsigned           loff = (unsigned)(size_t)(lds_bf16_t*)lptr;   // LDS byte offset
    unsigned long long ga   = (unsigned long long)(size_t)gptr;
    asm volatile("global_load_async_to_lds_b128 %0, %1, off"
                 :: "v"(loff), "v"(ga) : "memory");
}
static __device__ __forceinline__ void wait_async0() {
    asm volatile("s_wait_asynccnt 0x0" ::: "memory");
}

// Row reductions inside 16-lane half-waves (C-layout rows live across 16 lanes).
static __device__ __forceinline__ float rowmax16(float v) {
    v = fmaxf(v, __shfl_xor(v, 1, 32));
    v = fmaxf(v, __shfl_xor(v, 2, 32));
    v = fmaxf(v, __shfl_xor(v, 4, 32));
    v = fmaxf(v, __shfl_xor(v, 8, 32));
    return v;
}
static __device__ __forceinline__ float rowsum16(float v) {
    v += __shfl_xor(v, 1, 32);
    v += __shfl_xor(v, 2, 32);
    v += __shfl_xor(v, 4, 32);
    v += __shfl_xor(v, 8, 32);
    return v;
}

// ============================================================================
// Kernel 1: fused QKV projection.  x (64 MB) is read ONCE (memory-bound term).
// Each block: 16 rows of x, all 128 head cols, all 3 matrices, 8 waves.
// Q is pre-scaled by 1/sqrt(HEAD).  Q,K stored row-major [ROWS][128];
// V stored TRANSPOSED per batch: Vt[b][head][seq], so the attention kernel can
// stage V tiles with contiguous b128 async copies (no LDS transpose).
// ============================================================================
__global__ __launch_bounds__(256, 2)
void qkv_proj_kernel(const float* __restrict__ x,
                     const float* __restrict__ Wq, const float* __restrict__ Wk,
                     const float* __restrict__ Wv,
                     const float* __restrict__ bq, const float* __restrict__ bk,
                     const float* __restrict__ bv,
                     bf16* __restrict__ Qh, bf16* __restrict__ Kh, bf16* __restrict__ Vt,
                     float qscale)
{
    __shared__ __align__(16) bf16 xs[16][40];          // 16 rows x 32 k (padded)
    __shared__ __align__(16) bf16 wsT[3][128][40];     // transposed W tiles: [col][k]

    const int lane = threadIdx.x & 31;
    const int wave = threadIdx.x >> 5;
    const int lh   = lane & 15;
    const int sel  = lane >> 4;
    const int r0   = blockIdx.x * 16;                  // flattened B*T row tile
    const int n0   = wave * 16;                        // head-col tile

    const float* Ws[3] = {Wq, Wk, Wv};

    v8f acc[3];
    #pragma unroll
    for (int mt = 0; mt < 3; ++mt) acc[mt] = (v8f){0,0,0,0,0,0,0,0};

    for (int kb = 0; kb < EMBD; kb += 32) {
        // ---- cooperative stage: x tile (fp32 -> bf16) ----
        {
            int row = threadIdx.x >> 4;                // 0..15
            int col = (threadIdx.x & 15) * 2;          // 0..30
            const float* xp = x + (size_t)(r0 + row) * EMBD + kb + col;
            xs[row][col]     = (bf16)xp[0];
            xs[row][col + 1] = (bf16)xp[1];
            if (kb + 64 < EMBD)                        // prefetch x stream 2 steps ahead
                __builtin_prefetch((const void*)(xp + 64), 0, 3);
        }
        // ---- cooperative stage: W tiles, transposed for contiguous B-frags ----
        {
            int kq = threadIdx.x >> 3;                 // 0..31
            int cb = (threadIdx.x & 7) * 16;           // 0..112
            #pragma unroll
            for (int mt = 0; mt < 3; ++mt) {
                const float* wp = Ws[mt] + (size_t)(kb + kq) * HEADD + cb;
                #pragma unroll
                for (int j = 0; j < 16; ++j)
                    wsT[mt][cb + j][kq] = (bf16)wp[j];
            }
        }
        __syncthreads();

        // A fragment: rows of x (shared by all waves)
        v16bf afr = frag16(&xs[lh][sel * 8], &xs[lh][16 + sel * 8]);
        #pragma unroll
        for (int mt = 0; mt < 3; ++mt) {
            const bf16* wr = &wsT[mt][n0 + lh][sel * 8];
            v16bf bfr = frag16(wr, wr + 16);
            acc[mt] = wmma_bf16(afr, bfr, acc[mt]);
        }
        __syncthreads();
    }

    // ---- epilogue: bias, (Q) scale, bf16 stores ----
    const int col = n0 + lh;
    const float bqv = bq[col], bkv = bk[col], bvv = bv[col];
    #pragma unroll
    for (int i = 0; i < 8; ++i) {
        const int row = r0 + i + 8 * sel;              // flattened b*SEQ + t
        Qh[(size_t)row * HEADD + col] = (bf16)((acc[0][i] + bqv) * qscale);
        Kh[(size_t)row * HEADD + col] = (bf16)(acc[1][i] + bkv);
        const int bb = row >> 11;                      // row / SEQ
        const int tt = row & (SEQ - 1);                // row % SEQ
        Vt[((size_t)bb * HEADD + col) * SEQ + tt] = (bf16)(acc[2][i] + bvv);
    }
}

// ============================================================================
// Kernel 2: causal flash attention, one pass over keys, online softmax.
// Block = 4 waves = 64 query rows; wave owns a 16-row Q tile (Q frags in regs).
// K/V tiles DOUBLE-BUFFERED and staged with CDNA5 global_load_async_to_lds_b128:
// each iteration waits the tile issued last iteration, barriers once, launches
// the next tile's DMA, then overlaps 16 WMMAs + softmax with the DMA.
// ============================================================================
__global__ __launch_bounds__(128, 1)
void flash_attn_kernel(const bf16* __restrict__ Qh, const bf16* __restrict__ Kh,
                       const bf16* __restrict__ Vt, float* __restrict__ out)
{
    __shared__ __align__(16) bf16 ksm[2][32][136];     // K tiles row-major [key][head]
    __shared__ __align__(16) bf16 vsm[2][128][40];     // V tiles transposed [head][key]
    __shared__ __align__(16) bf16 psm[4][16][40];      // per-wave P relayout buffer

    const int lane = threadIdx.x & 31;
    const int wave = threadIdx.x >> 5;
    const int lh   = lane & 15;
    const int sel  = lane >> 4;
    const int b    = blockIdx.y;
    const int qblk = blockIdx.x * 64;
    const int qr0  = qblk + wave * 16;
    const size_t rowbase = (size_t)b * SEQ;

    // Stage one 32-key K/V tile into LDS buffer `bi` via async DMA (8 x b128/thread).
    auto stage = [&](int kb, int bi) {
        const int key = threadIdx.x >> 2;              // 0..31
        const int cb  = (threadIdx.x & 3) * 32;        // 0..96
        const bf16* kp = Kh + (rowbase + kb + key) * HEADD + cb;
        #pragma unroll
        for (int j = 0; j < 4; ++j)
            async_ld_b128(&ksm[bi][key][cb + j * 8], kp + j * 8);
        const bf16* vp = Vt + ((size_t)b * HEADD + threadIdx.x) * SEQ + kb;
        #pragma unroll
        for (int j = 0; j < 4; ++j)
            async_ld_b128(&vsm[bi][threadIdx.x][j * 8], vp + j * 8);
    };

    // ---- Q fragments (already scaled by 1/sqrt(H)); 4 K-steps over head dim ----
    v16bf qf[4];
    #pragma unroll
    for (int hi = 0; hi < 4; ++hi) {
        const bf16* qp = Qh + (rowbase + qr0 + lh) * HEADD + hi * 32;
        qf[hi] = frag16(qp + sel * 8, qp + 16 + sel * 8);
    }

    v8f o[8];
    float m[8], l[8];
    #pragma unroll
    for (int n = 0; n < 8; ++n) o[n] = (v8f){0,0,0,0,0,0,0,0};
    #pragma unroll
    for (int i = 0; i < 8; ++i) { m[i] = NEGBIG; l[i] = 0.0f; }

    const int kend = qblk + 64;                        // causal block bound
    stage(0, 0);                                       // prologue DMA
    int buf = 0;

    for (int kb = 0; kb < kend; kb += 32) {
        wait_async0();                                 // my share of tile `buf` is in LDS
        __syncthreads();                               // publish tile; alt buffer now free
        if (kb + 32 < kend) stage(kb + 32, buf ^ 1);   // DMA next tile under compute

        // ---- S = Q · K^T : two 16x16 key subtiles, K-dim = 128 ----
        v8f s[2];
        #pragma unroll
        for (int t = 0; t < 2; ++t) {
            v8f sc = (v8f){0,0,0,0,0,0,0,0};
            const bf16* kr = &ksm[buf][t * 16 + lh][0];
            #pragma unroll
            for (int hi = 0; hi < 4; ++hi) {
                v16bf bfr = frag16(kr + hi * 32 + sel * 8,
                                   kr + hi * 32 + sel * 8 + 16);
                sc = wmma_bf16(qf[hi], bfr, sc);
            }
            s[t] = sc;
        }

        // ---- causal mask + online softmax update ----
        #pragma unroll
        for (int i = 0; i < 8; ++i) {
            const int row = qr0 + i + 8 * sel;
            const int k0  = kb + lh;
            if (k0 > row)      s[0][i] = NEGBIG;
            if (k0 + 16 > row) s[1][i] = NEGBIG;
            float mx = rowmax16(fmaxf(s[0][i], s[1][i]));
            float mn = fmaxf(m[i], mx);
            float alpha = __expf(m[i] - mn);           // ==1 when both NEGBIG
            m[i] = mn;
            float p0 = __expf(s[0][i] - mn);
            float p1 = __expf(s[1][i] - mn);
            s[0][i] = p0; s[1][i] = p1;
            float rs = rowsum16(p0 + p1);
            l[i] = l[i] * alpha + rs;
            #pragma unroll
            for (int n = 0; n < 8; ++n) o[n][i] *= alpha;
        }

        // ---- P: C-layout f32 -> LDS -> A-layout bf16 fragment ----
        #pragma unroll
        for (int i = 0; i < 8; ++i) {
            psm[wave][i + 8 * sel][lh]      = (bf16)s[0][i];
            psm[wave][i + 8 * sel][16 + lh] = (bf16)s[1][i];
        }
        __builtin_amdgcn_wave_barrier();               // keep DS in program order
        const bf16* pr = &psm[wave][lh][0];
        v16bf pf = frag16(pr + sel * 8, pr + 16 + sel * 8);

        // ---- O += P · V : 8 head tiles, K-dim = 32 keys ----
        #pragma unroll
        for (int n = 0; n < 8; ++n) {
            const bf16* vr = &vsm[buf][n * 16 + lh][sel * 8];
            v16bf vf = frag16(vr, vr + 16);
            o[n] = wmma_bf16(pf, vf, o[n]);
        }
        buf ^= 1;
    }

    // ---- epilogue: normalize, fp32 store ----
    float inv[8];
    #pragma unroll
    for (int i = 0; i < 8; ++i) inv[i] = 1.0f / l[i];
    #pragma unroll
    for (int n = 0; n < 8; ++n) {
        #pragma unroll
        for (int i = 0; i < 8; ++i) {
            int row = qr0 + i + 8 * sel;
            out[(rowbase + row) * HEADD + n * 16 + lh] = o[n][i] * inv[i];
        }
    }
}

// ============================================================================
extern "C" void kernel_launch(void* const* d_in, const int* in_sizes, int n_in,
                              void* d_out, int out_size, void* d_ws, size_t ws_size,
                              hipStream_t stream) {
    const float* x  = (const float*)d_in[0];
    const float* Wq = (const float*)d_in[1];
    const float* Wk = (const float*)d_in[2];
    const float* Wv = (const float*)d_in[3];
    const float* bq = (const float*)d_in[4];
    const float* bk = (const float*)d_in[5];
    const float* bv = (const float*)d_in[6];

    // Workspace: bf16 Q, K (row-major) and Vt (per-batch transposed), 12.6 MB.
    bf16* Qh = (bf16*)d_ws;
    bf16* Kh = Qh + (size_t)ROWS * HEADD;
    bf16* Vt = Kh + (size_t)ROWS * HEADD;

    const float qscale = 0.08838834764831845f;         // 1/sqrt(128)

    qkv_proj_kernel<<<dim3(ROWS / 16), 256, 0, stream>>>(
        x, Wq, Wk, Wv, bq, bk, bv, Qh, Kh, Vt, qscale);

    flash_attn_kernel<<<dim3(SEQ / 64, BATCH), 128, 0, stream>>>(
        Qh, Kh, Vt, (float*)d_out);
}
